// TGNN_34557306863658
// MI455X (gfx1250) — compile-verified, matrix operand order checked by reading, not run on hardware
//
#include <hip/hip_runtime.h>

#define DIMQ   128
#define FEATQ  165
#define NROWS  262144
#define DINQ   549
#define NK1    18          // 18 * 32 = 576 padded K for GEMM1
#define NK2    4           // 4 * 32 = 128 K for GEMM2
#define ZP     584         // z / W1T pitch (bf16 elems), padded for LDS banks
#define W2P    136         // W2T / h pitch (bf16 elems)
#define PLNP   132         // pre-LN pitch (f32 elems)
#define ROWS   32          // rows per tile (two 16-row WMMA slabs per wave)
#define TPB    256
#define EPSQ   1e-5f

typedef __attribute__((ext_vector_type(16))) __bf16   v16bf;
typedef __attribute__((ext_vector_type(8)))  float    v8f;
typedef __attribute__((ext_vector_type(4)))  unsigned v4u;

union FragU { v16bf v; v4u q[2]; };

// ---- LDS carve-out (bytes) ----
#define OFF_W1T 0
#define SZ_W1T  (128 * ZP * 2)            // 149504
#define OFF_W2T (OFF_W1T + SZ_W1T)
#define SZ_W2T  (128 * W2P * 2)           // 34816
#define OFF_ZU  (OFF_W2T + SZ_W2T)
#define SZ_Z    (ROWS * ZP * 2)           // 37376
#define OFF_ZV  (OFF_ZU + SZ_Z)
#define OFF_HB  (OFF_ZV + SZ_Z)
#define SZ_HB   (ROWS * W2P * 2)          // 8704
#define OFF_PLN (OFF_HB + SZ_HB)
#define SZ_PLN  (ROWS * PLNP * 4)         // 16896
#define SMEM_BYTES (OFF_PLN + SZ_PLN)     // 284672 < 320KB WGP LDS

__device__ __forceinline__ unsigned pack2bf(float lo, float hi) {
    union { __bf16 h[2]; unsigned u; } p;
    p.h[0] = (__bf16)lo;                   // native f32->bf16 (v_cvt_pk path)
    p.h[1] = (__bf16)hi;
    return p.u;
}

__global__ __launch_bounds__(TPB, 1)
void tgnn_wmma_kernel(const float* __restrict__ hu, const float* __restrict__ hv,
                      const float* __restrict__ nu, const float* __restrict__ nv,
                      const float* __restrict__ x,
                      const float* __restrict__ W1, const float* __restrict__ b1,
                      const float* __restrict__ W2, const float* __restrict__ b2,
                      const float* __restrict__ gmm, const float* __restrict__ bta,
                      float* __restrict__ out)
{
    extern __shared__ char smem[];
    __bf16* w1t = (__bf16*)(smem + OFF_W1T);
    __bf16* w2t = (__bf16*)(smem + OFF_W2T);
    __bf16* zu  = (__bf16*)(smem + OFF_ZU);
    __bf16* zv  = (__bf16*)(smem + OFF_ZV);
    __bf16* hb  = (__bf16*)(smem + OFF_HB);
    float*  pln = (float*)(smem + OFF_PLN);
    unsigned* zu32 = (unsigned*)zu;
    unsigned* zv32 = (unsigned*)zv;

    const int tid  = threadIdx.x;
    const int lane = tid & 31;
    const int wv   = tid >> 5;          // n-tile id 0..7 (16 output cols each)
    const int lr   = lane & 15;
    const int hi   = lane >> 4;         // k-half selector for fragments

    // ---- one-time per-block: W1^T, W2^T into LDS as bf16 (zero-padded K) ----
    // thread pair-packs two consecutive k for one n; consecutive tids -> consecutive n
    for (int i = tid; i < (ZP / 2) * 128; i += TPB) {
        int n = i & 127, k2 = (i >> 7) * 2;
        float w0 = (k2     < DINQ) ? W1[(size_t)k2 * DIMQ + n]       : 0.f;
        float w1 = (k2 + 1 < DINQ) ? W1[(size_t)(k2 + 1) * DIMQ + n] : 0.f;
        ((unsigned*)w1t)[n * (ZP / 2) + (k2 >> 1)] = pack2bf(w0, w1);
    }
    for (int i = tid; i < (W2P / 2) * 128; i += TPB) {
        int n = i & 127, k2 = (i >> 7) * 2;
        float w0 = (k2     < DIMQ) ? W2[(size_t)k2 * DIMQ + n]       : 0.f;
        float w1 = (k2 + 1 < DIMQ) ? W2[(size_t)(k2 + 1) * DIMQ + n] : 0.f;
        ((unsigned*)w2t)[n * (W2P / 2) + (k2 >> 1)] = pack2bf(w0, w1);
    }

    const float b1v  = b1[wv * 16 + lr];
    const float b2v  = b2[wv * 16 + lr];
    const int   ocol = wv * 16 + lr;     // this lane's output column in C/D layout

    const float4* hu4 = (const float4*)hu;
    const float4* hv4 = (const float4*)hv;
    const float4* nu4 = (const float4*)nu;
    const float4* nv4 = (const float4*)nv;

    const int nTiles = NROWS / ROWS;     // 8192
    for (int tile = blockIdx.x; tile < nTiles; tile += gridDim.x) {
        const int row0 = tile * ROWS;
        __syncthreads();  // weights ready (iter 0) / prior-iter z reads done

        // ---- stage zu / zv tiles (bf16, pair-packed b32 stores) ----
        for (int i = tid; i < ROWS * (DIMQ / 4); i += TPB) {   // float4 loads
            int r = i >> 5, c4 = (i & 31) << 2;
            size_t g4 = ((size_t)(row0 + r) * DIMQ + c4) >> 2;
            float4 a = hu4[g4], b = hv4[g4], cu = nu4[g4], cv = nv4[g4];
            unsigned* zur = zu32 + r * (ZP / 2);
            unsigned* zvr = zv32 + r * (ZP / 2);
            int e = c4 >> 1;
            zur[e]            = pack2bf(a.x, a.y);  zur[e + 1]            = pack2bf(a.z, a.w);
            zur[e + 64]       = pack2bf(b.x, b.y);  zur[e + 65]           = pack2bf(b.z, b.w);
            zur[e + 128]      = pack2bf(cu.x, cu.y); zur[e + 129]         = pack2bf(cu.z, cu.w);
            zvr[e]            = pack2bf(b.x, b.y);  zvr[e + 1]            = pack2bf(b.z, b.w);
            zvr[e + 64]       = pack2bf(a.x, a.y);  zvr[e + 65]           = pack2bf(a.z, a.w);
            zvr[e + 128]      = pack2bf(cv.x, cv.y); zvr[e + 129]         = pack2bf(cv.z, cv.w);
        }
        for (int i = tid; i < ROWS * 100; i += TPB) {   // x + zero pad: cols 384..583
            int r = i / 100, p = (i % 100) * 2;
            float x0 = (p     < FEATQ) ? x[(size_t)(row0 + r) * FEATQ + p]     : 0.f;
            float x1 = (p + 1 < FEATQ) ? x[(size_t)(row0 + r) * FEATQ + p + 1] : 0.f;
            unsigned w = pack2bf(x0, x1);
            int e = (384 + p) >> 1;
            zu32[r * (ZP / 2) + e] = w;
            zv32[r * (ZP / 2) + e] = w;
        }

        // ---- prefetch next grid-stride tile into cache (global_prefetch_b8) ----
        {
            int tile2 = tile + gridDim.x;
            if (tile2 < nTiles) {
                int r2 = tile2 * ROWS;
                for (int i = tid; i < ROWS * 4; i += TPB) {      // 512B row = 4x128B
                    int r = i >> 2, c = (i & 3) * 32;
                    size_t g = (size_t)(r2 + r) * DIMQ + c;
                    __builtin_prefetch(hu + g, 0, 1);
                    __builtin_prefetch(hv + g, 0, 1);
                    __builtin_prefetch(nu + g, 0, 1);
                    __builtin_prefetch(nv + g, 0, 1);
                }
                for (int i = tid; i < ROWS * 6; i += TPB) {      // 660B row ~ 6 lines
                    int r = i / 6, c = (i % 6) * 32;             // c<=160 < FEAT
                    __builtin_prefetch(x + (size_t)(r2 + r) * FEATQ + c, 0, 1);
                }
            }
        }
        __syncthreads();

        for (int br = 0; br < 2; ++br) {
            const __bf16* z     = br ? zv : zu;
            const float* resid  = br ? hv : hu;
            float* obase = out + (size_t)br * (size_t)NROWS * DIMQ;

            // ---- GEMM1: h = relu(z @ W1 + b1); two 16-row slabs share B frag ----
            v8f acc0, acc1;
            #pragma unroll
            for (int i = 0; i < 8; ++i) { acc0[i] = b1v; acc1[i] = b1v; }

            const __bf16* a0   = z   + lr * ZP;
            const __bf16* a1   = z   + (16 + lr) * ZP;
            const __bf16* brow = w1t + (wv * 16 + lr) * ZP;
            #pragma unroll 3
            for (int s = 0; s < NK1; ++s) {
                const int kb = s * 32 + hi * 8;
                FragU A0, A1, B;
                B.q[0]  = *(const v4u*)(brow + kb);
                B.q[1]  = *(const v4u*)(brow + kb + 16);
                A0.q[0] = *(const v4u*)(a0 + kb);
                A0.q[1] = *(const v4u*)(a0 + kb + 16);
                A1.q[0] = *(const v4u*)(a1 + kb);
                A1.q[1] = *(const v4u*)(a1 + kb + 16);
                acc0 = __builtin_amdgcn_wmma_f32_16x16x32_bf16(
                         false, A0.v, false, B.v, (short)0, acc0, false, false);
                acc1 = __builtin_amdgcn_wmma_f32_16x16x32_bf16(
                         false, A1.v, false, B.v, (short)0, acc1, false, false);
            }
            #pragma unroll
            for (int i = 0; i < 8; ++i) {               // relu -> bf16 h tile
                float h0 = acc0[i] > 0.f ? acc0[i] : 0.f;
                float h1 = acc1[i] > 0.f ? acc1[i] : 0.f;
                hb[(i + hi * 8) * W2P + ocol]        = (__bf16)h0;
                hb[(16 + i + hi * 8) * W2P + ocol]   = (__bf16)h1;
            }
            __syncthreads();

            // ---- GEMM2: o = h @ W2 + b2 (dual slabs, shared B) ----
            v8f o0, o1;
            #pragma unroll
            for (int i = 0; i < 8; ++i) { o0[i] = b2v; o1[i] = b2v; }
            const __bf16* h0r  = hb  + lr * W2P;
            const __bf16* h1r  = hb  + (16 + lr) * W2P;
            const __bf16* b2r  = w2t + (wv * 16 + lr) * W2P;
            #pragma unroll
            for (int s = 0; s < NK2; ++s) {
                const int kb = s * 32 + hi * 8;
                FragU A0, A1, B;
                B.q[0]  = *(const v4u*)(b2r + kb);
                B.q[1]  = *(const v4u*)(b2r + kb + 16);
                A0.q[0] = *(const v4u*)(h0r + kb);
                A0.q[1] = *(const v4u*)(h0r + kb + 16);
                A1.q[0] = *(const v4u*)(h1r + kb);
                A1.q[1] = *(const v4u*)(h1r + kb + 16);
                o0 = __builtin_amdgcn_wmma_f32_16x16x32_bf16(
                       false, A0.v, false, B.v, (short)0, o0, false, false);
                o1 = __builtin_amdgcn_wmma_f32_16x16x32_bf16(
                       false, A1.v, false, B.v, (short)0, o1, false, false);
            }
            // residual (f32 from global) + stage pre-LN tile
            #pragma unroll
            for (int i = 0; i < 8; ++i) {
                int r = i + hi * 8;
                pln[r * PLNP + ocol] =
                    o0[i] + resid[(size_t)(row0 + r) * DIMQ + ocol];
                pln[(16 + r) * PLNP + ocol] =
                    o1[i] + resid[(size_t)(row0 + 16 + r) * DIMQ + ocol];
            }
            __syncthreads();

            // ---- LayerNorm: one row per 16-lane group, 2 passes for 32 rows ----
            #pragma unroll
            for (int pass = 0; pass < 2; ++pass) {
                int r = (tid >> 4) + pass * 16;          // 0..31
                int g = tid & 15;
                float s = 0.f, sq = 0.f, vals[8];
                #pragma unroll
                for (int j = 0; j < 8; ++j) {
                    float v = pln[r * PLNP + g + j * 16];
                    vals[j] = v; s += v; sq += v * v;
                }
                #pragma unroll
                for (int m = 1; m < 16; m <<= 1) {       // stays inside half-wave
                    s  += __shfl_xor(s,  m, 32);
                    sq += __shfl_xor(sq, m, 32);
                }
                float mean = s * (1.f / DIMQ);
                float var  = sq * (1.f / DIMQ) - mean * mean;
                float rstd = rsqrtf(var + EPSQ);
                float* orow = obase + (size_t)(row0 + r) * DIMQ;
                #pragma unroll
                for (int j = 0; j < 8; ++j) {
                    int c = g + j * 16;
                    orow[c] = (vals[j] - mean) * rstd * gmm[c] + bta[c];
                }
            }
            __syncthreads();   // protect hb/pln reuse and zu/zv restaging
        }
    }
}

extern "C" void kernel_launch(void* const* d_in, const int* in_sizes, int n_in,
                              void* d_out, int out_size, void* d_ws, size_t ws_size,
                              hipStream_t stream) {
    (void)in_sizes; (void)n_in; (void)d_ws; (void)ws_size; (void)out_size;
    const float* hu  = (const float*)d_in[0];
    const float* hv  = (const float*)d_in[1];
    const float* nu  = (const float*)d_in[2];
    const float* nv  = (const float*)d_in[3];
    const float* x   = (const float*)d_in[4];
    const float* W1  = (const float*)d_in[5];
    const float* b1  = (const float*)d_in[6];
    const float* W2  = (const float*)d_in[7];
    const float* b2  = (const float*)d_in[8];
    const float* gmm = (const float*)d_in[9];
    const float* bta = (const float*)d_in[10];

    (void)hipFuncSetAttribute((const void*)tgnn_wmma_kernel,
                              hipFuncAttributeMaxDynamicSharedMemorySize,
                              SMEM_BYTES);

    dim3 grid(2048), block(TPB);   // 8192 tiles, 4 per block (grid-stride)
    tgnn_wmma_kernel<<<grid, block, SMEM_BYTES, stream>>>(
        hu, hv, nu, nv, x, W1, b1, W2, b2, gmm, bta, (float*)d_out);
}
